// AttnBlock_8177617732199
// MI455X (gfx1250) — matrix operand order repaired and checked by reading, not compiled, not run on hardware
//
#include <hip/hip_runtime.h>

typedef __attribute__((ext_vector_type(16))) _Float16 v16h;
typedef __attribute__((ext_vector_type(8)))  _Float16 v8h;
typedef __attribute__((ext_vector_type(4)))  _Float16 v4h;
typedef __attribute__((ext_vector_type(8)))  float    v8f;

#define DEV_INLINE __device__ __forceinline__

static constexpr int Bn  = 2;
static constexpr int Ln  = 2048;   // sequence length (power of 2: shift 11)
static constexpr int Dn  = 1024;
static constexpr int Hn  = 16;
static constexpr int DHn = 64;

DEV_INLINE v16h cat16(v8h a, v8h b) {
    v16h r;
#pragma unroll
    for (int j = 0; j < 8; ++j) { r[j] = a[j]; r[j + 8] = b[j]; }
    return r;
}

// ---------------------------------------------------------------------------
// Kernel 1: LayerNorm (f32 in -> f16 out). One wave32 per row of 1024.
// ---------------------------------------------------------------------------
__global__ __launch_bounds__(256) void ln_f16_kernel(
    const float* __restrict__ x, const float* __restrict__ g,
    const float* __restrict__ b, _Float16* __restrict__ out) {
    const int row  = blockIdx.x * 8 + (threadIdx.x >> 5);
    const int lane = threadIdx.x & 31;
    const float4* xr = (const float4*)(x + (size_t)row * Dn);

    float4 v[8];
    float sum = 0.f, sq = 0.f;
#pragma unroll
    for (int i = 0; i < 8; ++i) {
        v[i] = xr[i * 32 + lane];
        sum += v[i].x + v[i].y + v[i].z + v[i].w;
        sq  += v[i].x * v[i].x + v[i].y * v[i].y + v[i].z * v[i].z + v[i].w * v[i].w;
    }
#pragma unroll
    for (int off = 16; off; off >>= 1) {
        sum += __shfl_xor(sum, off, 32);
        sq  += __shfl_xor(sq,  off, 32);
    }
    const float mu = sum * (1.f / Dn);
    const float rs = rsqrtf(sq * (1.f / Dn) - mu * mu + 1e-5f);

    const float4* gr = (const float4*)g;
    const float4* br = (const float4*)b;
    v4h* orow = (v4h*)(out + (size_t)row * Dn);
#pragma unroll
    for (int i = 0; i < 8; ++i) {
        const int idx = i * 32 + lane;
        float4 gg = gr[idx], bb = br[idx];
        v4h o;
        o[0] = (_Float16)((v[i].x - mu) * rs * gg.x + bb.x);
        o[1] = (_Float16)((v[i].y - mu) * rs * gg.y + bb.y);
        o[2] = (_Float16)((v[i].z - mu) * rs * gg.z + bb.z);
        o[3] = (_Float16)((v[i].w - mu) * rs * gg.w + bb.w);
        orow[idx] = o;
    }
}

// ---------------------------------------------------------------------------
// Kernel 2: f32 weight [K][N] -> f16 transposed [N][K].  Hoists the WMMA
// B-operand transpose out of the GEMM hot loop (fragments become contiguous).
// ---------------------------------------------------------------------------
__global__ __launch_bounds__(256) void cvt_wT_kernel(
    const float* __restrict__ w, _Float16* __restrict__ wt) {
    const int idx = blockIdx.x * blockDim.x + threadIdx.x;  // 0 .. 256K-1
    const int k  = idx >> 8;            // source row (K dim)
    const int n4 = (idx & 255) * 4;     // source cols
    float4 v = *(const float4*)(w + (size_t)k * Dn + n4);
    wt[(size_t)(n4 + 0) * Dn + k] = (_Float16)v.x;
    wt[(size_t)(n4 + 1) * Dn + k] = (_Float16)v.y;
    wt[(size_t)(n4 + 2) * Dn + k] = (_Float16)v.z;
    wt[(size_t)(n4 + 3) * Dn + k] = (_Float16)v.w;
}

// ---------------------------------------------------------------------------
// Kernel 3: LDS-free WMMA GEMM  C[M,N] = A[M,K] * Wt[N,K]^T + bias.
//   Each wave owns a 32x64 tile: 8 accumulators, 8 WMMAs per 32-deep k-step.
//   All fragments are per-lane 16B reads at k 8h.. and 16+8h.. (2x global_load_b128).
//   STORE_VT scatters the result as [B][H][DH][L] (pre-transposed V).
// ---------------------------------------------------------------------------
template <typename OutT, bool STORE_VT>
__global__ __launch_bounds__(256) void gemm_wmma_kernel(
    const _Float16* __restrict__ A, const _Float16* __restrict__ Wt,
    const float* __restrict__ bias, OutT* __restrict__ C,
    int M, int N, int K) {
    const int wave = threadIdx.x >> 5;
    const int lane = threadIdx.x & 31;
    const int ml   = lane & 15;
    const int hi   = lane >> 4;
    const int m0   = (blockIdx.x * 8 + wave) * 32;
    const int n0   = blockIdx.y * 64;

    v8f acc[2][4];
    const v8f vz = {};
#pragma unroll
    for (int mt = 0; mt < 2; ++mt)
#pragma unroll
        for (int t = 0; t < 4; ++t) acc[mt][t] = vz;

    for (int k0 = 0; k0 < K; k0 += 32) {
        if (k0 + 32 < K) {  // prefetch next k-tile (global_prefetch_b8)
            __builtin_prefetch(A  + (size_t)(m0 + lane) * K + k0 + 32, 0, 3);
            __builtin_prefetch(Wt + (size_t)(n0 + lane) * K + k0 + 32, 0, 3);
            __builtin_prefetch(Wt + (size_t)(n0 + 32 + lane) * K + k0 + 32, 0, 3);
        }
        v16h af[2];
#pragma unroll
        for (int mt = 0; mt < 2; ++mt) {
            const _Float16* pa = A + (size_t)(m0 + mt * 16 + ml) * K + k0 + hi * 8;
            af[mt] = cat16(*(const v8h*)pa, *(const v8h*)(pa + 16));
        }
#pragma unroll
        for (int t = 0; t < 4; ++t) {
            const _Float16* pb = Wt + (size_t)(n0 + t * 16 + ml) * K + k0 + hi * 8;
            v16h bf = cat16(*(const v8h*)pb, *(const v8h*)(pb + 16));
            acc[0][t] = __builtin_amdgcn_wmma_f32_16x16x32_f16(
                false, af[0], false, bf, (short)0, acc[0][t], false, false);
            acc[1][t] = __builtin_amdgcn_wmma_f32_16x16x32_f16(
                false, af[1], false, bf, (short)0, acc[1][t], false, false);
        }
    }

#pragma unroll
    for (int mt = 0; mt < 2; ++mt) {
#pragma unroll
        for (int t = 0; t < 4; ++t) {
            const int col = n0 + t * 16 + ml;
            const float bv = bias[col];
#pragma unroll
            for (int r = 0; r < 8; ++r) {
                const int row = m0 + mt * 16 + r + hi * 8;
                const float val = acc[mt][t][r] + bv;
                if (STORE_VT) {
                    // V^T layout: [B][H][DH][L]
                    const int bb = row >> 11, l = row & (Ln - 1);
                    const int hh = col >> 6,  dh = col & (DHn - 1);
                    C[(((size_t)(bb * Hn + hh) * DHn + dh) << 11) + l] = (OutT)val;
                } else {
                    C[(size_t)row * N + col] = (OutT)val;
                }
            }
        }
    }
}

// ---------------------------------------------------------------------------
// Kernel 4: causal flash attention. One wave per 16-row query tile,
// KV chunks of 32, online softmax. V comes in pre-transposed [B][H][DH][L],
// so every WMMA operand is two 16B row reads; only P bounces via LDS.
// ---------------------------------------------------------------------------
__global__ __launch_bounds__(256) void attn_wmma_kernel(
    const _Float16* __restrict__ Q, const _Float16* __restrict__ Km,
    const _Float16* __restrict__ Vt, _Float16* __restrict__ Oatt) {
    constexpr int PST = 40;  // P-row stride in halves (16B-aligned rows)
    __shared__ __attribute__((aligned(16))) _Float16 smem[8 * 16 * PST];

    const int tid  = threadIdx.x;
    const int wave = tid >> 5;
    const int lane = tid & 31;
    const int ml   = lane & 15;
    const int hi   = lane >> 4;

    _Float16* Pw = smem + wave * 16 * PST;

    const int bb = blockIdx.y >> 4;     // batch
    const int hh = blockIdx.y & 15;     // head
    const int qbase = (blockIdx.x * 8 + wave) * 16;

    const _Float16* Qh = Q  + ((size_t)bb * Ln) * Dn + hh * DHn;
    const _Float16* Kh = Km + ((size_t)bb * Ln) * Dn + hh * DHn;
    const _Float16* Vh = Vt + ((size_t)(bb * Hn + hh) * DHn) * Ln;

    // Q fragments: two 16x32 A-operands covering dh 0..31, 32..63
    v16h aq[2];
#pragma unroll
    for (int kb = 0; kb < 2; ++kb) {
        const _Float16* qp = Qh + (size_t)(qbase + ml) * Dn + kb * 32 + hi * 8;
        aq[kb] = cat16(*(const v8h*)qp, *(const v8h*)(qp + 16));
    }

    v8f o[4];
    const v8f vz = {};
#pragma unroll
    for (int t = 0; t < 4; ++t) o[t] = vz;
    float mr[8], lr[8];
#pragma unroll
    for (int r = 0; r < 8; ++r) { mr[r] = -__builtin_inff(); lr[r] = 0.f; }

    const int lastc = (qbase + 15) >> 5;
    for (int c = 0; c <= lastc; ++c) {
        const int kv0 = c * 32;

        if (c < lastc) {  // prefetch next chunk's K rows and V^T segments
            __builtin_prefetch(Kh + (size_t)(kv0 + 32 + lane) * Dn, 0, 3);
            __builtin_prefetch(Vh + (size_t)lane * Ln + kv0 + 32, 0, 3);
            __builtin_prefetch(Vh + (size_t)(32 + lane) * Ln + kv0 + 32, 0, 3);
        }

        // scores: S[nt] = Q * K^T for kv columns [kv0+16nt, kv0+16nt+16)
        v8f s[2];
#pragma unroll
        for (int nt = 0; nt < 2; ++nt) {
            s[nt] = vz;
#pragma unroll
            for (int kb = 0; kb < 2; ++kb) {
                const _Float16* kp =
                    Kh + (size_t)(kv0 + nt * 16 + ml) * Dn + kb * 32 + hi * 8;
                v16h bf = cat16(*(const v8h*)kp, *(const v8h*)(kp + 16));
                s[nt] = __builtin_amdgcn_wmma_f32_16x16x32_f16(
                    false, aq[kb], false, bf, (short)0, s[nt], false, false);
            }
        }

        // scale + causal mask (analytic; mask input unused)
        const bool needmask = (kv0 + 31) > qbase;
#pragma unroll
        for (int nt = 0; nt < 2; ++nt) {
#pragma unroll
            for (int r = 0; r < 8; ++r) {
                float v = s[nt][r] * 0.125f;  // 1/sqrt(64)
                if (needmask) {
                    const int col = kv0 + nt * 16 + ml;
                    const int row = qbase + r + hi * 8;
                    if (col > row) v = -__builtin_inff();
                }
                s[nt][r] = v;
            }
        }

        // online softmax (rows live across 16 lanes of each half-wave)
#pragma unroll
        for (int r = 0; r < 8; ++r) {
            float mx = fmaxf(s[0][r], s[1][r]);
#pragma unroll
            for (int off = 8; off; off >>= 1) mx = fmaxf(mx, __shfl_xor(mx, off, 16));
            const float mn = fmaxf(mr[r], mx);
            const float al = __expf(mr[r] - mn);
            const float p0 = __expf(s[0][r] - mn);
            const float p1 = __expf(s[1][r] - mn);
            float rs = p0 + p1;
#pragma unroll
            for (int off = 8; off; off >>= 1) rs += __shfl_xor(rs, off, 16);
            lr[r] = lr[r] * al + rs;
            mr[r] = mn;
            s[0][r] = p0; s[1][r] = p1;
#pragma unroll
            for (int t = 0; t < 4; ++t) o[t][r] *= al;
        }

        // P (C-layout) -> LDS -> A-fragment layout (the one unavoidable bounce)
#pragma unroll
        for (int r = 0; r < 8; ++r) {
            const int prow = r + hi * 8;
            Pw[prow * PST + ml]      = (_Float16)s[0][r];
            Pw[prow * PST + 16 + ml] = (_Float16)s[1][r];
        }
        const _Float16* pp = Pw + ml * PST;
        v16h ap = cat16(*(const v8h*)(pp + hi * 8), *(const v8h*)(pp + 16 + hi * 8));

        // O += P * V   (V^T rows are contiguous along kv)
#pragma unroll
        for (int t = 0; t < 4; ++t) {
            const _Float16* vp = Vh + (size_t)(t * 16 + ml) * Ln + kv0 + hi * 8;
            v16h bf = cat16(*(const v8h*)vp, *(const v8h*)(vp + 16));
            o[t] = __builtin_amdgcn_wmma_f32_16x16x32_f16(
                false, ap, false, bf, (short)0, o[t], false, false);
        }
    }

    // normalize and store heads back as [B, L, D] f16
    _Float16* ob = Oatt + ((size_t)bb * Ln) * Dn + hh * DHn;
#pragma unroll
    for (int r = 0; r < 8; ++r) {
        const float inv = 1.f / lr[r];
        const int row = qbase + r + hi * 8;
#pragma unroll
        for (int t = 0; t < 4; ++t)
            ob[(size_t)row * Dn + t * 16 + ml] = (_Float16)(o[t][r] * inv);
    }
}

// ---------------------------------------------------------------------------
// Host launcher
// ---------------------------------------------------------------------------
extern "C" void kernel_launch(void* const* d_in, const int* in_sizes, int n_in,
                              void* d_out, int out_size, void* d_ws, size_t ws_size,
                              hipStream_t stream) {
    (void)in_sizes; (void)n_in; (void)out_size; (void)ws_size;

    const float* x    = (const float*)d_in[0];
    /* d_in[1] = mask: causality computed analytically */
    const float* ln_g = (const float*)d_in[2];
    const float* ln_b = (const float*)d_in[3];
    const float* Wq   = (const float*)d_in[4];
    const float* bq   = (const float*)d_in[5];
    const float* Wk   = (const float*)d_in[6];
    const float* bk   = (const float*)d_in[7];
    const float* Wv   = (const float*)d_in[8];
    const float* bv   = (const float*)d_in[9];
    const float* Wo   = (const float*)d_in[10];
    const float* bo   = (const float*)d_in[11];
    float* out        = (float*)d_out;

    constexpr size_t M  = (size_t)Bn * Ln;   // 4096
    constexpr size_t MD = M * Dn;            // 4 Mi elements
    constexpr size_t DD = (size_t)Dn * Dn;   // 1 Mi elements

    _Float16* h16   = (_Float16*)d_ws;       // ws layout (f16): 48 MiB total
    _Float16* wqT   = h16  + MD;
    _Float16* wkT   = wqT  + DD;
    _Float16* wvT   = wkT  + DD;
    _Float16* woT   = wvT  + DD;
    _Float16* q16   = woT  + DD;
    _Float16* k16   = q16  + MD;
    _Float16* vt16  = k16  + MD;             // [B][H][DH][L]
    _Float16* att16 = vt16 + MD;

    // 1) LayerNorm -> f16
    ln_f16_kernel<<<M / 8, 256, 0, stream>>>(x, ln_g, ln_b, h16);

    // 2) weights -> f16, transposed [N][K]
    const int cvtBlocks = (int)(DD / 4 / 256);
    cvt_wT_kernel<<<cvtBlocks, 256, 0, stream>>>(Wq, wqT);
    cvt_wT_kernel<<<cvtBlocks, 256, 0, stream>>>(Wk, wkT);
    cvt_wT_kernel<<<cvtBlocks, 256, 0, stream>>>(Wv, wvT);
    cvt_wT_kernel<<<cvtBlocks, 256, 0, stream>>>(Wo, woT);

    // 3) Q/K/V projections (LDS-free WMMA GEMMs); V stored pre-transposed
    dim3 ggrid((unsigned)(M / 32 / 8), Dn / 64);
    gemm_wmma_kernel<_Float16, false><<<ggrid, 256, 0, stream>>>(
        h16, wqT, bq, q16, (int)M, Dn, Dn);
    gemm_wmma_kernel<_Float16, false><<<ggrid, 256, 0, stream>>>(
        h16, wkT, bk, k16, (int)M, Dn, Dn);
    gemm_wmma_kernel<_Float16, true><<<ggrid, 256, 0, stream>>>(
        h16, wvT, bv, vt16, (int)M, Dn, Dn);

    // 4) causal flash attention (WMMA), heads merged back to [B, L, D] f16
    dim3 agrid((Ln / 16) / 8, Bn * Hn);
    attn_wmma_kernel<<<agrid, 256, 0, stream>>>(q16, k16, vt16, att16);

    // 5) output projection (WMMA GEMM, f32 out into d_out)
    gemm_wmma_kernel<float, false><<<ggrid, 256, 0, stream>>>(
        att16, woT, bo, out, (int)M, Dn, Dn);
}
